// Discriminator_4672924418481
// MI455X (gfx1250) — matrix operand order repaired and checked by reading, not compiled, not run on hardware
//
#include <hip/hip_runtime.h>
#include <hip/hip_bf16.h>
#include <cstdint>
#include <cstddef>

typedef __attribute__((ext_vector_type(16))) _Float16 v16h;
typedef __attribute__((ext_vector_type(8)))  _Float16 v8h;
typedef __attribute__((ext_vector_type(8)))  float    v8f;

static constexpr int kB    = 256;
static constexpr int kT    = 512;
static constexpr int kHD   = 256;
static constexpr int kG    = 4 * kHD;      // 1024 gate columns
static constexpr int kFEAT = 48;
static constexpr int kKPAD = 64;
static constexpr int kM    = kB * kT;      // 131072 activation rows
static constexpr float kSlope = 0.1f;

__device__ __forceinline__ v8f vz8() {
  v8f z;
#pragma unroll
  for (int e = 0; e < 8; ++e) z[e] = 0.f;
  return z;
}
__device__ __forceinline__ v16h cat16(v8h lo, v8h hi) {
  return __builtin_shufflevector(lo, hi, 0,1,2,3,4,5,6,7,8,9,10,11,12,13,14,15);
}
__device__ __forceinline__ v16h ld16(const _Float16* p) {  // two b128 loads
  return cat16(*(const v8h*)p, *(const v8h*)(p + 8));
}
__device__ __forceinline__ float sigf(float x) { return 1.f / (1.f + __expf(-x)); }

// ---------------- prep kernels ----------------

// Composite first-layer weight: W_a[64][256]; rows 0-7 = w1 real rows,
// rows 8..47 = emb_w folded through w1, rows 48..63 = 0 (K padding).
__global__ void make_wa(const float* __restrict__ w1,   // (40,256)
                        const float* __restrict__ emb,  // (4,10,8)
                        float* __restrict__ wa) {       // (64,256)
  int idx = blockIdx.x * 256 + threadIdx.x;
  if (idx >= kKPAD * kHD) return;
  int h = idx & 255, r = idx >> 8;
  float v = 0.f;
  if (r < 8) {
    v = w1[r * kHD + h];
  } else if (r < kFEAT) {
    int c = (r - 8) / 10, n = (r - 8) % 10;
#pragma unroll
    for (int e = 0; e < 8; ++e)
      v += emb[(c * 10 + n) * 8 + e] * w1[(8 + c * 8 + e) * kHD + h];
  }
  wa[idx] = v;
}

// Pack a (Ksrc x N) row-major f32 weight into WMMA B-fragment order (f16):
// dst[(((ntile*KIT_total + kloc)*32) + lane)*16 + j]
//   = W[kloc*32 + (lane<16?0:16) + j][ntile*16 + (lane&15)]
__global__ void pack_b(const float* __restrict__ W, _Float16* __restrict__ dst,
                       int Ksrc, int N, int KIT_total, int kit_off) {
  int idx = blockIdx.x * 256 + threadIdx.x;
  if (idx >= Ksrc * N) return;
  int j    = idx & 15;
  int lane = (idx >> 4) & 31;
  int rest = idx >> 9;
  int kitsrc = Ksrc >> 5;
  int kloc  = rest % kitsrc;
  int ntile = rest / kitsrc;
  int col  = ntile * 16 + (lane & 15);
  int krow = kloc * 32 + ((lane < 16) ? 0 : 16) + j;
  _Float16 v = (_Float16)W[(size_t)krow * N + col];
  dst[(((size_t)(ntile * KIT_total + kloc + kit_off) * 32) + lane) * 16 + j] = v;
}

// Pad x0 (B*T,48) f32 -> (B*T,64) f16 (zero-padded K)
__global__ void pad_x0(const float* __restrict__ x0, _Float16* __restrict__ xh) {
  int idx = blockIdx.x * 256 + threadIdx.x;
  if (idx >= kM * kKPAD) return;
  int c = idx & 63, row = idx >> 6;
  float v = (c < kFEAT) ? x0[(size_t)row * kFEAT + c] : 0.f;
  xh[idx] = (_Float16)v;
}

// ---------------- WMMA GEMM + bias + leaky-relu (templated K) ----------------
// One wave computes a 16x64 output tile (4 N-tiles); base+immediate addressing;
// A and B fragments double-buffered to avoid WMMA WAR hazard NOPs.
template <int KIT>
__global__ __launch_bounds__(32) void gemm_lrelu_t(
    const _Float16* __restrict__ A, const _Float16* __restrict__ Bp,
    const float* __restrict__ bias, _Float16* __restrict__ C, int N) {
  int mtile = blockIdx.x;
  int nbase = blockIdx.y * 4;
  int lane  = threadIdx.x;

  v8f acc[4];
#pragma unroll
  for (int n4 = 0; n4 < 4; ++n4) acc[n4] = vz8();

  int rowA = mtile * 16 + (lane & 15);
  int koff = (lane < 16) ? 0 : 16;
  const _Float16* arow  = A + (size_t)rowA * (KIT * 32) + koff;
  const _Float16* bbase = Bp + (size_t)nbase * (KIT * 512) + (size_t)lane * 16;

  v16h Af = ld16(arow);
  v16h Bf[4], Bn[4];
#pragma unroll
  for (int n4 = 0; n4 < 4; ++n4) Bf[n4] = ld16(bbase + (n4 * KIT) * 512);

#pragma unroll
  for (int k = 0; k < KIT; ++k) {
    v16h Afn;
    if (k + 1 < KIT) {
      Afn = ld16(arow + (k + 1) * 32);
#pragma unroll
      for (int n4 = 0; n4 < 4; ++n4)
        Bn[n4] = ld16(bbase + (n4 * KIT + k + 1) * 512);
    }
#pragma unroll
    for (int n4 = 0; n4 < 4; ++n4)
      acc[n4] = __builtin_amdgcn_wmma_f32_16x16x32_f16(
          false, Af, false, Bf[n4], (short)0, acc[n4], false, false);
    if (k + 1 < KIT) {
      Af = Afn;
#pragma unroll
      for (int n4 = 0; n4 < 4; ++n4) Bf[n4] = Bn[n4];
    }
  }

#pragma unroll
  for (int n4 = 0; n4 < 4; ++n4) {
    int col = (nbase + n4) * 16 + (lane & 15);
    float bv = bias[col];
#pragma unroll
    for (int r = 0; r < 8; ++r) {
      float v = acc[n4][r] + bv;
      v = (v >= 0.f) ? v : kSlope * v;
      int m = mtile * 16 + ((lane < 16) ? r : r + 8);
      C[(size_t)m * N + col] = (_Float16)v;
    }
  }
}

// ---------------- gate pre-activation GEMM ----------------
// gx2[t][gatecol][b] = x3[b*T+t,:] @ wi[:,gatecol] + bi[gatecol]   (f16)
// Time-major + batch-innermost so the recurrence pulls each lane's 8 D-tile
// seed values as one contiguous 16B load.
__global__ __launch_bounds__(32) void gemm_gx(
    const _Float16* __restrict__ A,   // (B*T, 256) f16
    const _Float16* __restrict__ Bp,  // packed wi, KIT=8, 64 ntiles
    const float* __restrict__ bias,   // bi (1024)
    _Float16* __restrict__ gx2) {     // (T, 1024, B) f16
  constexpr int KIT = 8;
  int mtile = blockIdx.x;
  int nbase = blockIdx.y * 4;
  int lane  = threadIdx.x;

  v8f acc[4];
#pragma unroll
  for (int n4 = 0; n4 < 4; ++n4) acc[n4] = vz8();

  int rowA = mtile * 16 + (lane & 15);
  int koff = (lane < 16) ? 0 : 16;
  const _Float16* arow  = A + (size_t)rowA * kHD + koff;
  const _Float16* bbase = Bp + (size_t)nbase * (KIT * 512) + (size_t)lane * 16;

  v16h Af = ld16(arow);
  v16h Bf[4], Bn[4];
#pragma unroll
  for (int n4 = 0; n4 < 4; ++n4) Bf[n4] = ld16(bbase + (n4 * KIT) * 512);

#pragma unroll
  for (int k = 0; k < KIT; ++k) {
    v16h Afn;
    if (k + 1 < KIT) {
      Afn = ld16(arow + (k + 1) * 32);
#pragma unroll
      for (int n4 = 0; n4 < 4; ++n4)
        Bn[n4] = ld16(bbase + (n4 * KIT + k + 1) * 512);
    }
#pragma unroll
    for (int n4 = 0; n4 < 4; ++n4)
      acc[n4] = __builtin_amdgcn_wmma_f32_16x16x32_f16(
          false, Af, false, Bf[n4], (short)0, acc[n4], false, false);
    if (k + 1 < KIT) {
      Af = Afn;
#pragma unroll
      for (int n4 = 0; n4 < 4; ++n4) Bf[n4] = Bn[n4];
    }
  }

#pragma unroll
  for (int n4 = 0; n4 < 4; ++n4) {
    int col = (nbase + n4) * 16 + (lane & 15);
    float bv = bias[col];
#pragma unroll
    for (int r = 0; r < 8; ++r) {
      int m = mtile * 16 + ((lane < 16) ? r : r + 8);
      int b = m >> 9;        // T = 512
      int t = m & (kT - 1);
      gx2[((size_t)t * kG + col) * kB + b] = (_Float16)(acc[n4][r] + bv);
    }
  }
}

// ---------------- fused LSTM recurrence ----------------
// 32 blocks: 0..15 forward, 16..31 reverse; each owns 16 batch rows for all
// 512 steps.  8 waves/block; wave w owns hidden units [w*32, w*32+32).
// Per step: gates(16x1024) = gx2[t] (seeds) + q(16x256) @ wh.
// wh k=0..3 is register-resident across the whole time loop; k=4..7 streams
// from L2.  All fragment loads are base + immediate-offset.
__global__ __launch_bounds__(256) void lstm_kernel(
    const _Float16* __restrict__ gxF,  // (T,1024,B) f16, bi folded in
    const _Float16* __restrict__ gxR,
    const _Float16* __restrict__ whF,  // packed wh, KIT=8, 64 ntiles
    const _Float16* __restrict__ whR,
    const float* __restrict__ bhF,     // bh (1024)
    const float* __restrict__ bhR,
    float* __restrict__ qF,            // (B,256) f32 final hidden fwd
    float* __restrict__ qR) {
  __shared__ __align__(16) _Float16 qs[16][kHD];  // q_{t-1}

  int blk   = blockIdx.x;
  int dir   = blk >> 4;
  int btile = blk & 15;
  int tid   = threadIdx.x;
  int lane  = tid & 31;
  int wv    = tid >> 5;

  const _Float16* gx = dir ? gxR : gxF;
  const _Float16* wp = dir ? whR : whF;
  const float*    bh = dir ? bhR : bhF;
  float*          qo = dir ? qR : qF;

  for (int i = tid; i < 16 * kHD; i += 256) (&qs[0][0])[i] = (_Float16)0.f;

  float bg[4][2];
#pragma unroll
  for (int qd = 0; qd < 4; ++qd)
#pragma unroll
    for (int hf = 0; hf < 2; ++hf)
      bg[qd][hf] = bh[qd * kHD + wv * 32 + hf * 16 + (lane & 15)];

  v8f sst[2];
  sst[0] = vz8();
  sst[1] = vz8();

  int rowA = lane & 15;
  int koff = (lane < 16) ? 0 : 16;
  int b0   = btile * 16 + ((lane < 16) ? 0 : 8);
  int ldrow = tid >> 4, ldcol = (tid & 15) << 4;

  // base pointers: everything else is an immediate offset
  // weight frag (tile qd,hf ; k): halfs off = qd*65536 + hf*4096 + k*512
  const _Float16* wb = wp + (size_t)wv * 8192 + (size_t)lane * 16;
  // seed (tile qd,hf): halfs off = qd*65536 + hf*4096 (per time slice)
  const _Float16* sb = gx + (size_t)wv * 8192 + (size_t)(lane & 15) * 256 + b0;
  // A frag (k): halfs off = k*32
  const _Float16* qbase = &qs[rowA][koff];

  // register-resident wh for k = 0..3 (32 fragments, held across time loop)
  v16h Wreg[8][4];
#pragma unroll
  for (int t8 = 0; t8 < 8; ++t8)
#pragma unroll
    for (int k = 0; k < 4; ++k)
      Wreg[t8][k] = ld16(wb + (t8 >> 1) * 65536 + (t8 & 1) * 4096 + k * 512);

  for (int t = 0; t < kT; ++t) {
    __syncthreads();  // prev qs writes complete
    int tt = dir ? (kT - 1 - t) : t;
    const _Float16* sbt = sb + (size_t)tt * (kG * kB);

    // issue gate seed loads early; consumed only after the WMMA loop
    v8h sd[8];
#pragma unroll
    for (int t8 = 0; t8 < 8; ++t8)
      sd[t8] = *(const v8h*)(sbt + (t8 >> 1) * 65536 + (t8 & 1) * 4096);

    if (t + 1 < kT) {  // prefetch next step's seed lines (global_prefetch_b8)
      int tn = dir ? (kT - 2 - t) : (t + 1);
      const _Float16* sbn = sb + (size_t)tn * (kG * kB);
#pragma unroll
      for (int t8 = 0; t8 < 8; ++t8)
        __builtin_prefetch(sbn + (t8 >> 1) * 65536 + (t8 & 1) * 4096, 0, 1);
    }

    v8f acc[8];
#pragma unroll
    for (int i = 0; i < 8; ++i) acc[i] = vz8();

    v16h Af = ld16(qbase);
#pragma unroll
    for (int k = 0; k < 8; ++k) {
      v16h Bs[8];
      if (k >= 4) {   // streamed half of wh
#pragma unroll
        for (int t8 = 0; t8 < 8; ++t8)
          Bs[t8] = ld16(wb + (t8 >> 1) * 65536 + (t8 & 1) * 4096 + k * 512);
      }
      v16h Afn;
      if (k < 7) Afn = ld16(qbase + (k + 1) * 32);
#pragma unroll
      for (int t8 = 0; t8 < 8; ++t8)
        acc[t8] = __builtin_amdgcn_wmma_f32_16x16x32_f16(
            false, Af, false, (k < 4) ? Wreg[t8][k] : Bs[t8],
            (short)0, acc[t8], false, false);
      if (k < 7) Af = Afn;
    }
    __syncthreads();  // all qs reads done before rewrite

    // gate nonlinearity + state update in native D-matrix layout
#pragma unroll
    for (int hf = 0; hf < 2; ++hf) {
      int col = wv * 32 + hf * 16 + (lane & 15);
#pragma unroll
      for (int r = 0; r < 8; ++r) {
        float fg = acc[0 + hf][r] + (float)sd[0 + hf][r] + bg[0][hf];
        float ig = acc[2 + hf][r] + (float)sd[2 + hf][r] + bg[1][hf];
        float ag = acc[4 + hf][r] + (float)sd[4 + hf][r] + bg[2][hf];
        float og = acc[6 + hf][r] + (float)sd[6 + hf][r] + bg[3][hf];
        float sn = sigf(fg) * sst[hf][r] + sigf(ig) * tanhf(ag);
        float qv = sigf(og) * tanhf(sn);
        sst[hf][r] = sn;
        int m = (lane < 16) ? r : r + 8;
        qs[m][col] = (_Float16)qv;
      }
    }
  }
  __syncthreads();
  {   // emit final hidden state (f32)
    float* dst = qo + (size_t)(btile * 16 + ldrow) * kHD + ldcol;
#pragma unroll
    for (int j = 0; j < 16; ++j) dst[j] = (float)qs[ldrow][ldcol + j];
  }
}

// ---------------- output head ----------------
__global__ __launch_bounds__(256) void final_out(
    const float* __restrict__ qf, const float* __restrict__ qr,
    const float* __restrict__ w3, const float* __restrict__ b3,
    float* __restrict__ out) {
  int b = threadIdx.x;  // B == 256
  float acc = b3[0];
  for (int j = 0; j < kHD; ++j)
    acc += qf[(size_t)b * kHD + j] * w3[j] + qr[(size_t)b * kHD + j] * w3[kHD + j];
  out[b] = acc;
}

// ---------------- host launcher ----------------
static inline size_t alignup(size_t x) { return (x + 255) & ~(size_t)255; }

extern "C" void kernel_launch(void* const* d_in, const int* in_sizes, int n_in,
                              void* d_out, int out_size, void* d_ws, size_t ws_size,
                              hipStream_t stream) {
  const float* x0    = (const float*)d_in[0];
  const float* emb_w = (const float*)d_in[1];
  const float* w1    = (const float*)d_in[2];
  const float* b1    = (const float*)d_in[3];
  const float* w2    = (const float*)d_in[4];
  const float* b2    = (const float*)d_in[5];
  const float* wi_f  = (const float*)d_in[6];
  const float* bi_f  = (const float*)d_in[7];
  const float* wh_f  = (const float*)d_in[8];
  const float* bh_f  = (const float*)d_in[9];
  const float* wi_r  = (const float*)d_in[10];
  const float* bi_r  = (const float*)d_in[11];
  const float* wh_r  = (const float*)d_in[12];
  const float* bh_r  = (const float*)d_in[13];
  const float* w3    = (const float*)d_in[14];
  const float* b3    = (const float*)d_in[15];
  float* out = (float*)d_out;

  char* p = (char*)d_ws;
  size_t off = 0;
  auto take = [&](size_t bytes) { char* r = p + off; off = alignup(off + bytes); return r; };

  _Float16* x0h   = (_Float16*)take((size_t)kM * kKPAD * 2);
  _Float16* x2h   = (_Float16*)take((size_t)kM * kHD * 2);
  _Float16* x3h   = (_Float16*)take((size_t)kM * kHD * 2);
  _Float16* gxF   = (_Float16*)take((size_t)kT * kG * kB * 2);   // 268 MB
  _Float16* gxR   = (_Float16*)take((size_t)kT * kG * kB * 2);   // 268 MB
  float*    wa    = (float*)   take((size_t)kKPAD * kHD * 4);
  _Float16* pkWa  = (_Float16*)take((size_t)16 * 2 * 32 * 16 * 2);
  _Float16* pkW2  = (_Float16*)take((size_t)16 * 8 * 32 * 16 * 2);
  _Float16* pkWiF = (_Float16*)take((size_t)64 * 8 * 32 * 16 * 2);
  _Float16* pkWiR = (_Float16*)take((size_t)64 * 8 * 32 * 16 * 2);
  _Float16* pkWhF = (_Float16*)take((size_t)64 * 8 * 32 * 16 * 2);
  _Float16* pkWhR = (_Float16*)take((size_t)64 * 8 * 32 * 16 * 2);
  float*    qf    = (float*)   take((size_t)kB * kHD * 4);
  float*    qr    = (float*)   take((size_t)kB * kHD * 4);
  (void)ws_size; (void)n_in; (void)in_sizes; (void)out_size;

  // ---- weight prep ----
  make_wa<<<(kKPAD * kHD + 255) / 256, 256, 0, stream>>>(w1, emb_w, wa);
  pack_b<<<(kKPAD * kHD + 255) / 256, 256, 0, stream>>>(wa, pkWa, kKPAD, kHD, 2, 0);
  pack_b<<<(kHD * kHD + 255) / 256, 256, 0, stream>>>(w2, pkW2, kHD, kHD, 8, 0);
  pack_b<<<(kHD * kG + 255) / 256, 256, 0, stream>>>(wi_f, pkWiF, kHD, kG, 8, 0);
  pack_b<<<(kHD * kG + 255) / 256, 256, 0, stream>>>(wi_r, pkWiR, kHD, kG, 8, 0);
  pack_b<<<(kHD * kG + 255) / 256, 256, 0, stream>>>(wh_f, pkWhF, kHD, kG, 8, 0);
  pack_b<<<(kHD * kG + 255) / 256, 256, 0, stream>>>(wh_r, pkWhR, kHD, kG, 8, 0);

  // ---- activation pipeline ----
  pad_x0<<<(kM * kKPAD + 255) / 256, 256, 0, stream>>>(x0, x0h);

  dim3 g1(kM / 16, kHD / 64);
  gemm_lrelu_t<2><<<g1, 32, 0, stream>>>(x0h, pkWa, b1, x2h, kHD);  // K=64
  gemm_lrelu_t<8><<<g1, 32, 0, stream>>>(x2h, pkW2, b2, x3h, kHD);  // K=256

  // ---- gate pre-activations (parallel; hoisted out of the recurrence) ----
  dim3 g2(kM / 16, kG / 64);
  gemm_gx<<<g2, 32, 0, stream>>>(x3h, pkWiF, bi_f, gxF);
  gemm_gx<<<g2, 32, 0, stream>>>(x3h, pkWiR, bi_r, gxR);

  // ---- recurrence: 16 batch tiles x 2 directions ----
  lstm_kernel<<<32, 256, 0, stream>>>(gxF, gxR, pkWhF, pkWhR, bh_f, bh_r, qf, qr);

  // ---- head ----
  final_out<<<1, 256, 0, stream>>>(qf, qr, w3, b3, out);
}
